// GraphConvolutionalLayer_30958124270375
// MI455X (gfx1250) — matrix operand order repaired and checked by reading, not compiled, not run on hardware
//
#include <hip/hip_runtime.h>

// ---------------------------------------------------------------------------
// GCN layer for MI455X (gfx1250, wave32, WMMA bf16, async LDS copies).
//   y = D^-1/2 X (bf16, pre-transposed)            [prep]
//   support = D^-1/2 (adj @ y + y)                 [GEMM1: 8192x8192x512]
//   out = relu(support @ W^T + b)                  [GEMM2: 8192x512x512]
// adj (256 MB fp32) streams HBM twice (rowsum + GEMM A); same-row-panel
// blocks are adjacent in launch order so the 192 MB L2 absorbs the 4x
// column-block reuse. Ping-pong LDS, one barrier per k-step. bf16 operand
// tiles are moved with GLOBAL_LOAD_ASYNC_TO_LDS_B128 (ASYNCcnt), the fp32
// adj tile goes through VGPRs for the v_cvt_pk_bf16_f32 conversion.
// ---------------------------------------------------------------------------

#define N_NODES 8192
#define DIM_IN  512
#define DIM_OUT 512

typedef __attribute__((ext_vector_type(16))) __bf16          v16bf;
typedef __attribute__((ext_vector_type(2)))  __bf16          v2bf;
typedef __attribute__((ext_vector_type(8)))  float           v8f;
typedef __attribute__((ext_vector_type(2)))  float           v2f;
typedef __attribute__((ext_vector_type(16))) unsigned short  v16u;
typedef __attribute__((ext_vector_type(8)))  unsigned short  v8u;

static __device__ __forceinline__ unsigned short f2bf(float f) {
    __bf16 h = (__bf16)f;
    return __builtin_bit_cast(unsigned short, h);
}
// two floats -> packed bf16 pair (single v_cvt_pk_bf16_f32, no fixups)
static __device__ __forceinline__ unsigned int packbf2(float a, float b) {
    v2f f; f.x = a; f.y = b;
    return __builtin_bit_cast(unsigned int, __builtin_convertvector(f, v2bf));
}
static __device__ __forceinline__ float bf2f(unsigned short u) {
    unsigned int w = ((unsigned int)u) << 16;
    return __builtin_bit_cast(float, w);
}

// ---- CDNA5 async global->LDS copy (16 bytes per lane, ASYNCcnt) ----------
static __device__ __forceinline__ void
async_copy_b128(const void* gsrc, void* lds_dst) {
    // generic pointer to __shared__: low 32 bits are the LDS byte address
    unsigned loff = (unsigned)(unsigned long long)lds_dst;
    unsigned long long ga = (unsigned long long)gsrc;
    asm volatile("global_load_async_to_lds_b128 %0, %1, off"
                 :: "v"(loff), "v"(ga) : "memory");
}
static __device__ __forceinline__ void wait_async0() {
#if __has_builtin(__builtin_amdgcn_s_wait_asynccnt)
    __builtin_amdgcn_s_wait_asynccnt(0);
#else
    asm volatile("s_wait_asynccnt 0x0" ::: "memory");
#endif
}

// ---------------------------------------------------------------------------
// K1: d[row] = rsqrt(sum_j adj[row][j] + 1)   (one 256 MB streaming pass)
// ---------------------------------------------------------------------------
__global__ __launch_bounds__(256)
void gcn_rowsum_kernel(const float* __restrict__ adj, float* __restrict__ d) {
    __shared__ float red[256];
    const int row = blockIdx.x;
    const int t   = threadIdx.x;
    const float4* rp = (const float4*)(adj + (size_t)row * N_NODES);
    float s = 0.0f;
#pragma unroll
    for (int j = 0; j < N_NODES / (256 * 4); ++j) {   // 8 x float4 per thread
        const float4 v = rp[t + 256 * j];
        s += (v.x + v.y) + (v.z + v.w);
    }
    red[t] = s;
    __syncthreads();
    for (int off = 128; off > 0; off >>= 1) {
        if (t < off) red[t] += red[t + off];
        __syncthreads();
    }
    if (t == 0) d[row] = rsqrtf(red[0] + 1.0f);   // +1 from the identity
}

// ---------------------------------------------------------------------------
// K2: yT[c][r] = bf16(d[r] * x[r][c])  (LDS 32x32 tile transpose)
// ---------------------------------------------------------------------------
__global__ __launch_bounds__(256)
void gcn_scale_transpose_kernel(const float* __restrict__ x,
                                const float* __restrict__ d,
                                unsigned short* __restrict__ yT) {
    __shared__ float tile[32][33];
    const int rBase = blockIdx.x * 32;   // node rows
    const int cBase = blockIdx.y * 32;   // feature cols
#pragma unroll
    for (int i = 0; i < 4; ++i) {
        int r = threadIdx.y + 8 * i;
        tile[r][threadIdx.x] =
            x[(size_t)(rBase + r) * DIM_IN + cBase + threadIdx.x];
    }
    __syncthreads();
#pragma unroll
    for (int i = 0; i < 4; ++i) {
        int c = threadIdx.y + 8 * i;
        int r = threadIdx.x;
        float v = tile[r][c] * d[rBase + r];
        yT[(size_t)(cBase + c) * N_NODES + rBase + r] = f2bf(v);
    }
}

// ---------------------------------------------------------------------------
// K3: W -> bf16 (row-major W is the transposed-B layout for support @ W^T)
// ---------------------------------------------------------------------------
__global__ __launch_bounds__(256)
void gcn_w_bf16_kernel(const float* __restrict__ W, unsigned short* __restrict__ Wb) {
    int i = blockIdx.x * 256 + threadIdx.x;
    Wb[i] = f2bf(W[i]);
}

// ---------------------------------------------------------------------------
// K4: tiled bf16 WMMA GEMM. Block tile 128x128, 8 wave32 waves (4x2 grid),
// each wave owns 2x4 16x16 accumulators = 8 WMMA per k-step. Ping-pong LDS.
//
// MODE 0: A = adj fp32 (VGPR staging + cvt); epilogue:
//         support = bf16((acc + y[row][col]) * d[row])   (+y = folded +I)
// MODE 1: A = support bf16 (async copy);   epilogue:
//         out = relu(acc + bias[col]) fp32
// ---------------------------------------------------------------------------
template <int MODE>
__global__ __launch_bounds__(256)
void gcn_gemm_kernel(const void* __restrict__ Aptr,
                     const unsigned short* __restrict__ BT, int ldb, int K,
                     const float* __restrict__ d,
                     const unsigned short* __restrict__ yT,
                     const float* __restrict__ bias,
                     void* __restrict__ Cptr) {
    __shared__ __align__(16) unsigned short As[2][128 * 32];  // 2 x 8 KB
    __shared__ __align__(16) unsigned short Bs[2][128 * 32];  // 2 x 8 KB

    const int tid   = threadIdx.x;
    const int lane  = tid & 31;
    const int wave  = tid >> 5;
    const int waveM = wave & 3;            // 4 waves down   (32 rows each)
    const int waveN = wave >> 2;           // 2 waves across (64 cols each)
    // blockIdx.x = column block -> adjacent blocks share A rows (L2 reuse)
    const int rowBase = blockIdx.y * 128;
    const int colBase = blockIdx.x * 128;

    const int rowInLane = lane & 15;
    const int half      = lane >> 4;       // 0: lanes 0-15, 1: lanes 16-31
    const int kbaseA    = half * 16;       // A frag: K = kbase..kbase+15
    const int koffB     = half * 16;       // B frag: K = koff..koff+15

    v8f acc[2][4] = {};
    float4 fa[4];                          // MODE 0 A staging regs

    // per-thread staging coordinates
    const int bR = tid >> 2;               // 128-row index for 16B chunks
    const int bC = (tid & 3) << 3;         // bf16 column (x8)
    const int aR = tid >> 3;               // MODE 0: float4 row
    const int aC = (tid & 7) << 2;         // MODE 0: float4 col

    // ---- staging helpers -------------------------------------------------
    auto stageB_async = [&](int k0, unsigned short* buf) {
#pragma unroll
        for (int j = 0; j < 2; ++j) {
            int r = bR + 64 * j;
            async_copy_b128(BT + (size_t)(colBase + r) * ldb + k0 + bC,
                            &buf[r * 32 + bC]);
        }
    };
    auto stageA_async = [&](int k0, unsigned short* buf) {   // MODE 1 only
        const unsigned short* A = (const unsigned short*)Aptr;
#pragma unroll
        for (int j = 0; j < 2; ++j) {
            int r = bR + 64 * j;
            async_copy_b128(A + (size_t)(rowBase + r) * DIM_IN + k0 + bC,
                            &buf[r * 32 + bC]);
        }
    };
    auto loadA_regs = [&](int k0) {                          // MODE 0 only
        const float* A = (const float*)Aptr;
#pragma unroll
        for (int j = 0; j < 4; ++j) {
            int r = aR + 32 * j;
            const float* gp = A + (size_t)(rowBase + r) * N_NODES + k0 + aC;
            fa[j] = *(const float4*)gp;
            if (k0 + 64 < K)                 // global_prefetch_b8 (L2 warm)
                __builtin_prefetch(gp + 64, 0, 3);
        }
    };
    auto storeA_lds = [&](unsigned short* buf) {             // MODE 0 only
#pragma unroll
        for (int j = 0; j < 4; ++j) {
            int r = aR + 32 * j;
            uint2 pk;
            pk.x = packbf2(fa[j].x, fa[j].y);
            pk.y = packbf2(fa[j].z, fa[j].w);
            *(uint2*)(&buf[r * 32 + aC]) = pk;
        }
    };

    // ---- prologue: stage tile k = 0 --------------------------------------
    stageB_async(0, Bs[0]);
    if constexpr (MODE == 0) {
        loadA_regs(0);
        storeA_lds(As[0]);
    } else {
        stageA_async(0, As[0]);
    }
    wait_async0();
    __syncthreads();

    int p = 0;
    for (int k0 = 0; k0 < K; k0 += 32) {
        const bool more = (k0 + 32 < K);
        // ---- issue copies for tile k+1 into buffer p^1 (overlaps WMMAs) --
        if (more) {
            stageB_async(k0 + 32, Bs[p ^ 1]);
            if constexpr (MODE == 0) loadA_regs(k0 + 32);
            else                     stageA_async(k0 + 32, As[p ^ 1]);
        }

        // ---- fragments from LDS buffer p ----
        v16bf afrag[2];
#pragma unroll
        for (int tm = 0; tm < 2; ++tm) {
            const unsigned short* ab =
                &As[p][(waveM * 32 + tm * 16 + rowInLane) * 32];
            v8u lo = *(const v8u*)(ab + kbaseA);        // K = kbase..kbase+7
            v8u hi = *(const v8u*)(ab + kbaseA + 8);    // K = kbase+8..kbase+15
            v16u a;
#pragma unroll
            for (int i = 0; i < 8; ++i) { a[i] = lo[i]; a[8 + i] = hi[i]; }
            afrag[tm] = __builtin_bit_cast(v16bf, a);
        }
        v16bf bfrag[4];
#pragma unroll
        for (int tn = 0; tn < 4; ++tn) {
            const unsigned short* bb =
                &Bs[p][(waveN * 64 + tn * 16 + rowInLane) * 32 + koffB];
            bfrag[tn] = __builtin_bit_cast(v16bf, *(const v16u*)bb);
        }

#pragma unroll
        for (int tm = 0; tm < 2; ++tm)
#pragma unroll
            for (int tn = 0; tn < 4; ++tn)
                acc[tm][tn] = __builtin_amdgcn_wmma_f32_16x16x32_bf16(
                    false, afrag[tm], false, bfrag[tn],
                    (short)0, acc[tm][tn], false, false);

        // ---- commit converted A tile, then publish buffer p^1 ------------
        if (more) {
            if constexpr (MODE == 0) storeA_lds(As[p ^ 1]);
            wait_async0();
        }
        __syncthreads();
        p ^= 1;
    }

    // ---- epilogue (C layout: VGPR r -> M = r + 8*half, N = lane & 15) ----
    const int nlane = lane & 15;
    const int mbase = half * 8;
#pragma unroll
    for (int tm = 0; tm < 2; ++tm) {
#pragma unroll
        for (int tn = 0; tn < 4; ++tn) {
            const int col = colBase + waveN * 64 + tn * 16 + nlane;
#pragma unroll
            for (int r = 0; r < 8; ++r) {
                const int row = rowBase + waveM * 32 + tm * 16 + mbase + r;
                float v = acc[tm][tn][r];
                if constexpr (MODE == 0) {
                    // + y[row][col] is the folded identity term of a_hat
                    float yv = bf2f(yT[(size_t)col * N_NODES + row]);
                    ((unsigned short*)Cptr)[(size_t)row * DIM_IN + col] =
                        f2bf((v + yv) * d[row]);
                } else {
                    ((float*)Cptr)[(size_t)row * DIM_OUT + col] =
                        fmaxf(v + bias[col], 0.0f);
                }
            }
        }
    }
}

// ---------------------------------------------------------------------------
// launch: inputs in setup_inputs() order: x, adj, W, b  (all fp32)
// ---------------------------------------------------------------------------
extern "C" void kernel_launch(void* const* d_in, const int* in_sizes, int n_in,
                              void* d_out, int out_size, void* d_ws, size_t ws_size,
                              hipStream_t stream) {
    const float* x   = (const float*)d_in[0];
    const float* adj = (const float*)d_in[1];
    const float* W   = (const float*)d_in[2];
    const float* b   = (const float*)d_in[3];
    float* out = (float*)d_out;

    // workspace layout (~16.6 MB)
    char* ws = (char*)d_ws;
    float*          d_deg   = (float*)ws;                                   // 32 KB
    unsigned short* yT      = (unsigned short*)(ws + (32u << 10));          // 8 MB
    unsigned short* Wb      = (unsigned short*)(ws + (32u << 10) + (8u << 20)); // 512 KB
    unsigned short* support = (unsigned short*)(ws + (32u << 10) + (8u << 20) + (512u << 10)); // 8 MB

    // 1) degree^-1/2
    gcn_rowsum_kernel<<<N_NODES, 256, 0, stream>>>(adj, d_deg);
    // 2) yT = transpose(diag(d) * x) in bf16
    gcn_scale_transpose_kernel<<<dim3(N_NODES / 32, DIM_IN / 32), dim3(32, 8), 0, stream>>>(
        x, d_deg, yT);
    // 3) W -> bf16
    gcn_w_bf16_kernel<<<(DIM_OUT * DIM_IN) / 256, 256, 0, stream>>>(W, Wb);
    // 4) support = diag(d) * (adj @ y + y)      [8192 x 8192 x 512 bf16 WMMA]
    gcn_gemm_kernel<0><<<dim3(DIM_IN / 128, N_NODES / 128), 256, 0, stream>>>(
        (const void*)adj, yT, N_NODES, N_NODES, d_deg, yT, nullptr, (void*)support);
    // 5) out = relu(support @ W^T + b)          [8192 x 512 x 512 bf16 WMMA]
    gcn_gemm_kernel<1><<<dim3(DIM_OUT / 128, N_NODES / 128), 256, 0, stream>>>(
        (const void*)support, Wb, DIM_IN, DIM_IN, d_deg, nullptr, b, (void*)out);
}
